// CrossModalAttention_33466385170569
// MI455X (gfx1250) — matrix operand order repaired and checked by reading, not compiled, not run on hardware
//
#include <hip/hip_runtime.h>
#include <hip/hip_bf16.h>
#include <stdint.h>

// ---------------------------------------------------------------------------
// CDNA5 (gfx1250) cross-modal MHA, fully bf16-WMMA with async LDS staging.
//   pass 1: convert activations f32->bf16; convert+transpose weights -> [K][N]
//   pass 2: GEMMs (Q scaled-bf16, K head-transposed bf16, V bf16) -- all tile
//           staging via GLOBAL_LOAD_ASYNC_TO_LDS_B128, double-buffered, BK=64
//   pass 3: flash attention (bf16 in/out), async K/V staging, 128-key chunks
//   pass 4: output projection GEMM -> f32 + bias
// ---------------------------------------------------------------------------

typedef __attribute__((ext_vector_type(16))) __bf16 v16bf;
typedef __attribute__((ext_vector_type(8)))  __bf16 v8bf;
typedef __attribute__((ext_vector_type(4)))  __bf16 v4bf;
typedef __attribute__((ext_vector_type(2)))  __bf16 v2bf;
typedef __attribute__((ext_vector_type(8)))  float  v8f;
typedef __attribute__((ext_vector_type(4)))  float  v4f;

union Frag16 { v16bf v; v8bf h[2]; };

__device__ __forceinline__ v8f wmma_bf16(v16bf a, v16bf b, v8f c) {
  return __builtin_amdgcn_wmma_f32_16x16x32_bf16(false, a, false, b,
                                                 (short)0, c, false, false);
}

// Async global->LDS copy, 16 bytes per lane. Generic LDS pointer's low 32 bits
// are the LDS byte offset (ISA aperture rule), which is what VDST wants.
__device__ __forceinline__ void async_b128(void* lds, const void* g) {
  unsigned off = (unsigned)(uintptr_t)lds;
  asm volatile("global_load_async_to_lds_b128 %0, %1, off"
               :: "v"(off), "v"(g) : "memory");
}
__device__ __forceinline__ void wait_async0() {
  asm volatile("s_wait_asynccnt 0x0" ::: "memory");
}

// ---------------------------------------------------------------------------
// Pass-1 kernels: elementwise f32->bf16, and transpose-convert W[N][K]->Wt[K][N]
// ---------------------------------------------------------------------------
__global__ __launch_bounds__(256)
void cvt_kernel(const float* __restrict__ src, __bf16* __restrict__ dst, int n)
{
  int i = (blockIdx.x * 256 + threadIdx.x) * 4;
  if (i < n) {
    v4f s = *(const v4f*)&src[i];
    v4bf d = { (__bf16)s[0], (__bf16)s[1], (__bf16)s[2], (__bf16)s[3] };
    *(v4bf*)&dst[i] = d;
  }
}

__global__ __launch_bounds__(256)
void transpose_cvt_kernel(const float* __restrict__ W, __bf16* __restrict__ Wt,
                          int N, int K)
{
  __shared__ float tile[32][33];
  const int n0 = blockIdx.y * 32, k0 = blockIdx.x * 32;
  const int tx = threadIdx.x & 31, ty4 = (threadIdx.x >> 5) * 4;
#pragma unroll
  for (int i = 0; i < 4; ++i)
    tile[ty4 + i][tx] = W[(size_t)(n0 + ty4 + i) * K + k0 + tx];
  __syncthreads();
#pragma unroll
  for (int i = 0; i < 4; ++i)
    Wt[(size_t)(k0 + ty4 + i) * N + n0 + tx] = (__bf16)tile[tx][ty4 + i];
}

// ---------------------------------------------------------------------------
// GEMM: acc = A[M,K] @ Wt[K,N] (+bias), bf16 inputs, async double-buffered LDS.
// Tile 128x128, BK=64 (2 WMMA k-steps per stage -> 16 WMMAs per barrier pair).
// MODE 0: f32 out + bias            (output projection)
// MODE 1: bf16 out, (acc+bias)/8    (Q, scale folded)
// MODE 2: bf16 out transposed per-head -> Kt[col][Sdim]  (K)
// MODE 3: bf16 out + bias           (V)
// ---------------------------------------------------------------------------
template<int MODE>
__global__ __launch_bounds__(256)
void gemm_kernel(const __bf16* __restrict__ A, const __bf16* __restrict__ Wt,
                 const float* __restrict__ bias, void* __restrict__ outp,
                 int M, int N, int K, int Sdim)
{
  constexpr int BM = 128, BN = 128, BK = 64;
  __shared__ alignas(16) __bf16 As[2][BM][BK];   // [m][k], k contiguous (16KB x2)
  __shared__ alignas(16) __bf16 Bs[2][BK][BN];   // [k][n], n contiguous (16KB x2)

  const int tid  = threadIdx.x;
  const int lane = tid & 31;
  const int wave = tid >> 5;
  const int wm   = wave & 3;       // 4 x 32 rows
  const int wn   = wave >> 2;      // 2 x 64 cols
  const int m0   = blockIdx.y * BM;
  const int n0   = blockIdx.x * BN;
  const int half = lane >> 4;
  const int l16  = lane & 15;

  // 1024 A-chunks (128 rows x 8) + 1024 B-chunks (64 rows x 16), 16B each.
  auto stage = [&](int buf, int k0) {
#pragma unroll
    for (int i = 0; i < 4; ++i) {
      int c  = tid + i * 256;
      int ar = c >> 3, ae = (c & 7) * 8;
      async_b128(&As[buf][ar][ae], &A[(size_t)(m0 + ar) * K + k0 + ae]);
      int br = c >> 4, be = (c & 15) * 8;
      async_b128(&Bs[buf][br][be], &Wt[(size_t)(k0 + br) * N + n0 + be]);
    }
  };

  v8f acc[2][4];
#pragma unroll
  for (int mi = 0; mi < 2; ++mi)
#pragma unroll
    for (int ni = 0; ni < 4; ++ni)
#pragma unroll
      for (int r = 0; r < 8; ++r) acc[mi][ni][r] = 0.0f;

  stage(0, 0);
  wait_async0();
  __syncthreads();

  int buf = 0;
  for (int k0 = 0; k0 < K; k0 += BK) {
    if (k0 + BK < K) stage(buf ^ 1, k0 + BK);   // overlap copies with WMMA

#pragma unroll
    for (int kk = 0; kk < 2; ++kk) {
      Frag16 af[2];
      const int koff = kk * 32 + (half ? 8 : 0);
#pragma unroll
      for (int mi = 0; mi < 2; ++mi) {
        int row = wm * 32 + mi * 16 + l16;
        af[mi].h[0] = *(const v8bf*)&As[buf][row][koff];
        af[mi].h[1] = *(const v8bf*)&As[buf][row][koff + 16];
      }
#pragma unroll
      for (int ni = 0; ni < 4; ++ni) {
        Frag16 bf_;
#pragma unroll
        for (int s = 0; s < 16; ++s)
          bf_.v[s] = Bs[buf][kk * 32 + (s < 8 ? 0 : 16) + 8 * half + (s & 7)][wn * 64 + ni * 16 + l16];
#pragma unroll
        for (int mi = 0; mi < 2; ++mi)
          acc[mi][ni] = wmma_bf16(af[mi].v, bf_.v, acc[mi][ni]);
      }
    }
    wait_async0();
    __syncthreads();
    buf ^= 1;
  }

#pragma unroll
  for (int ni = 0; ni < 4; ++ni) {
    int col = n0 + wn * 64 + ni * 16 + l16;
    float bv = bias[col];
#pragma unroll
    for (int mi = 0; mi < 2; ++mi) {
      int rowb = m0 + wm * 32 + mi * 16 + 8 * half;
      if constexpr (MODE == 0) {
        float* C = (float*)outp;
#pragma unroll
        for (int r = 0; r < 8; ++r)
          C[(size_t)(rowb + r) * N + col] = acc[mi][ni][r] + bv;
      } else if constexpr (MODE == 1) {
        __bf16* C = (__bf16*)outp;
#pragma unroll
        for (int r = 0; r < 8; ++r)
          C[(size_t)(rowb + r) * N + col] =
              (__bf16)((acc[mi][ni][r] + bv) * 0.125f);
      } else if constexpr (MODE == 2) {
        // Kt[col][Sdim]: D rows r are consecutive s -> pack pairs.
        __bf16* Kt = (__bf16*)outp;
#pragma unroll
        for (int j = 0; j < 4; ++j) {
          v2bf p = { (__bf16)(acc[mi][ni][2 * j]     + bv),
                     (__bf16)(acc[mi][ni][2 * j + 1] + bv) };
          *(v2bf*)&Kt[(size_t)col * Sdim + rowb + 2 * j] = p;
        }
      } else {
        __bf16* C = (__bf16*)outp;
#pragma unroll
        for (int r = 0; r < 8; ++r)
          C[(size_t)(rowb + r) * N + col] = (__bf16)(acc[mi][ni][r] + bv);
      }
    }
  }
}

// ---------------------------------------------------------------------------
// Flash attention: one head per blockIdx.y, 128 queries per blockIdx.x.
// Q bf16 pre-scaled [N][1024]; Kt bf16 [1024][S] (head-transposed); V bf16.
// 128-key chunks, K/V double-buffered via async copies. bf16 output.
// LDS: Qs 16K + Kts 32K + Vs 32K + Ps 32K = 112 KB (<= 2 WGs / 320KB WGP).
// ---------------------------------------------------------------------------
__global__ __launch_bounds__(256)
void attn_kernel(const __bf16* __restrict__ Qg, const __bf16* __restrict__ Ktg,
                 const __bf16* __restrict__ Vg, __bf16* __restrict__ Og, int S)
{
  constexpr int BM = 128, BS = 128, D = 64, HD = 1024;
  constexpr int NT = BS / 16;                    // 8 score n-tiles
  __shared__ alignas(16) __bf16 Qs [BM][D];
  __shared__ alignas(16) __bf16 Kts[2][D][BS];   // [d][s]
  __shared__ alignas(16) __bf16 Vs [2][BS][D];   // [s][d]
  __shared__ alignas(16) __bf16 Ps [BM][BS];

  const int tid  = threadIdx.x;
  const int lane = tid & 31;
  const int wave = tid >> 5;
  const int half = lane >> 4;
  const int l16  = lane & 15;
  const int h    = blockIdx.y;
  const int row0 = blockIdx.x * BM;

  auto stage_kv = [&](int buf, int s0) {
#pragma unroll
    for (int i = 0; i < 4; ++i) {
      int c  = tid + i * 256;                    // 1024 chunks each
      int dd = c >> 4, ke = (c & 15) * 8;
      async_b128(&Kts[buf][dd][ke], &Ktg[((size_t)h * 64 + dd) * S + s0 + ke]);
      int s = c >> 3, ve = (c & 7) * 8;
      async_b128(&Vs[buf][s][ve],   &Vg[(size_t)(s0 + s) * HD + h * D + ve]);
    }
  };

  // stage Q once (async) together with first K/V tile
#pragma unroll
  for (int i = 0; i < 4; ++i) {
    int c = tid + i * 256;                       // 1024 chunks
    int r = c >> 3, e = (c & 7) * 8;
    async_b128(&Qs[r][e], &Qg[(size_t)(row0 + r) * HD + h * D + e]);
  }
  stage_kv(0, 0);
  wait_async0();
  __syncthreads();

  Frag16 qf[2];
  {
    int row = wave * 16 + l16;
#pragma unroll
    for (int kk = 0; kk < 2; ++kk) {
      int koff = kk * 32 + (half ? 8 : 0);
      qf[kk].h[0] = *(const v8bf*)&Qs[row][koff];
      qf[kk].h[1] = *(const v8bf*)&Qs[row][koff + 16];
    }
  }

  v8f o_[4];
  float m_[8], l_[8];
#pragma unroll
  for (int t = 0; t < 4; ++t)
#pragma unroll
    for (int r = 0; r < 8; ++r) o_[t][r] = 0.0f;
#pragma unroll
  for (int r = 0; r < 8; ++r) { m_[r] = -__builtin_inff(); l_[r] = 0.0f; }

  int buf = 0;
  for (int s0 = 0; s0 < S; s0 += BS) {
    if (s0 + BS < S) stage_kv(buf ^ 1, s0 + BS);

    // scores S[16x128] = Q(16xd) x Kt(dx128)
    v8f sacc[NT];
#pragma unroll
    for (int t = 0; t < NT; ++t)
#pragma unroll
      for (int r = 0; r < 8; ++r) sacc[t][r] = 0.0f;
#pragma unroll
    for (int kk = 0; kk < 2; ++kk)
#pragma unroll
      for (int t = 0; t < NT; ++t) {
        Frag16 kb;
#pragma unroll
        for (int s = 0; s < 16; ++s)
          kb.v[s] = Kts[buf][kk * 32 + (s < 8 ? 0 : 16) + 8 * half + (s & 7)][t * 16 + l16];
        sacc[t] = wmma_bf16(qf[kk].v, kb.v, sacc[t]);
      }

    // online softmax; rows r / r+8 live in complementary 16-lane halves
#pragma unroll
    for (int r = 0; r < 8; ++r) {
      float mx = sacc[0][r];
#pragma unroll
      for (int t = 1; t < NT; ++t) mx = fmaxf(mx, sacc[t][r]);
#pragma unroll
      for (int msk = 1; msk <= 8; msk <<= 1)
        mx = fmaxf(mx, __shfl_xor(mx, msk, 32));
      float mn    = fmaxf(m_[r], mx);
      float alpha = __expf(m_[r] - mn);
      m_[r] = mn;
      float rs = 0.0f;
#pragma unroll
      for (int t = 0; t < NT; ++t) {
        float p = __expf(sacc[t][r] - mn);
        sacc[t][r] = p;
        rs += p;
      }
#pragma unroll
      for (int msk = 1; msk <= 8; msk <<= 1)
        rs += __shfl_xor(rs, msk, 32);
      l_[r] = l_[r] * alpha + rs;
#pragma unroll
      for (int t = 0; t < 4; ++t) o_[t][r] *= alpha;
#pragma unroll
      for (int t = 0; t < NT; ++t)
        Ps[wave * 16 + r + 8 * half][t * 16 + l16] = (__bf16)sacc[t][r];
    }
    __syncthreads();

    // O += P(16x128) x V(128xd)
#pragma unroll
    for (int kk = 0; kk < 4; ++kk) {
      Frag16 pf;
      int prow = wave * 16 + l16;
      int koff = kk * 32 + (half ? 8 : 0);
      pf.h[0] = *(const v8bf*)&Ps[prow][koff];
      pf.h[1] = *(const v8bf*)&Ps[prow][koff + 16];
#pragma unroll
      for (int t = 0; t < 4; ++t) {
        Frag16 vb_;
#pragma unroll
        for (int s = 0; s < 16; ++s)
          vb_.v[s] = Vs[buf][kk * 32 + (s < 8 ? 0 : 16) + 8 * half + (s & 7)][t * 16 + l16];
        o_[t] = wmma_bf16(pf.v, vb_.v, o_[t]);
      }
    }
    wait_async0();
    __syncthreads();
    buf ^= 1;
  }

#pragma unroll
  for (int r = 0; r < 8; ++r) {
    float inv = 1.0f / l_[r];
    int row = row0 + wave * 16 + r + 8 * half;
#pragma unroll
    for (int t = 0; t < 4; ++t)
      Og[(size_t)row * HD + h * D + t * 16 + l16] = (__bf16)(o_[t][r] * inv);
  }
}

// ---------------------------------------------------------------------------
// Launch
// ---------------------------------------------------------------------------
extern "C" void kernel_launch(void* const* d_in, const int* in_sizes, int n_in,
                              void* d_out, int out_size, void* d_ws, size_t ws_size,
                              hipStream_t stream) {
  const float* graph = (const float*)d_in[0];   // [8192,1024]
  const float* other = (const float*)d_in[1];   // [4096,768]
  const float* Wq    = (const float*)d_in[2];   // [1024,1024]
  const float* bq    = (const float*)d_in[3];
  const float* Wk    = (const float*)d_in[4];   // [1024,768]
  const float* bk    = (const float*)d_in[5];
  const float* Wv    = (const float*)d_in[6];   // [1024,768]
  const float* bv    = (const float*)d_in[7];
  const float* Wo    = (const float*)d_in[8];   // [1024,1024]
  const float* bo    = (const float*)d_in[9];
  float* out = (float*)d_out;

  char* ws = (char*)d_ws;
  __bf16* Gb  = (__bf16*)(ws);                      // 16 MiB  graph bf16
  __bf16* Ob  = (__bf16*)(ws + 16777216ull);        //  6 MiB  other bf16
  __bf16* Wqt = (__bf16*)(ws + 23068672ull);        //  2 MiB  [1024][1024]
  __bf16* Wkt = (__bf16*)(ws + 25165824ull);        //  1.5MiB [768][1024]
  __bf16* Wvt = (__bf16*)(ws + 26738688ull);        //  1.5MiB [768][1024]
  __bf16* Wot = (__bf16*)(ws + 28311552ull);        //  2 MiB  [1024][1024]
  __bf16* Qb  = (__bf16*)(ws + 30408704ull);        // 16 MiB  Q bf16 (scaled)
  __bf16* Ktb = (__bf16*)(ws + 47185920ull);        //  8 MiB  Kt [1024][4096]
  __bf16* Vb  = (__bf16*)(ws + 55574528ull);        //  8 MiB  V bf16
  __bf16* Ab  = (__bf16*)(ws + 63963136ull);        // 16 MiB  attn out bf16

  dim3 blk(256);
  // pass 1: conversions
  cvt_kernel<<<8192, blk, 0, stream>>>(graph, Gb, 8192 * 1024);
  cvt_kernel<<<3072, blk, 0, stream>>>(other, Ob, 4096 * 768);
  transpose_cvt_kernel<<<dim3(32, 32), blk, 0, stream>>>(Wq, Wqt, 1024, 1024);
  transpose_cvt_kernel<<<dim3(24, 32), blk, 0, stream>>>(Wk, Wkt, 1024, 768);
  transpose_cvt_kernel<<<dim3(24, 32), blk, 0, stream>>>(Wv, Wvt, 1024, 768);
  transpose_cvt_kernel<<<dim3(32, 32), blk, 0, stream>>>(Wo, Wot, 1024, 1024);
  // pass 2: projections
  gemm_kernel<1><<<dim3(8, 64), blk, 0, stream>>>(Gb, Wqt, bq, Qb, 8192, 1024, 1024, 0);
  gemm_kernel<2><<<dim3(8, 32), blk, 0, stream>>>(Ob, Wkt, bk, Ktb, 4096, 1024, 768, 4096);
  gemm_kernel<3><<<dim3(8, 32), blk, 0, stream>>>(Ob, Wvt, bv, Vb, 4096, 1024, 768, 0);
  // pass 3: attention
  attn_kernel<<<dim3(64, 16), blk, 0, stream>>>(Qb, Ktb, Vb, Ab, 4096);
  // pass 4: output projection
  gemm_kernel<0><<<dim3(8, 64), blk, 0, stream>>>(Ab, Wot, bo, out, 8192, 1024, 1024, 0);
}
